// GraphRec_24833500905764
// MI455X (gfx1250) — matrix operand order repaired and checked by reading, not compiled, not run on hardware
//
#include <hip/hip_runtime.h>
#include <hip/hip_bf16.h>

typedef _Float16 half_t;
typedef __attribute__((ext_vector_type(16))) _Float16 v16h;
typedef __attribute__((ext_vector_type(8)))  _Float16 v8h;
typedef __attribute__((ext_vector_type(8)))  float    v8f;
typedef unsigned int v4u __attribute__((ext_vector_type(4)));
typedef int          v8i __attribute__((ext_vector_type(8)));
typedef int          v4i __attribute__((ext_vector_type(4)));

#define NTHR 128   // 4 wave32 per workgroup

// ---------------------------------------------------------------------------
// Fragment loaders per CDNA5 ISA 7.12.2 layouts (wave32).
// A: 16x32 (MxK). lanes 0-15: halves 0..7 = K k0..k0+7, halves 8..15 = k0+16..23
//                 lanes 16-31: halves 0..7 = k0+8..15, halves 8..15 = k0+24..31
__device__ inline v16h load_a_frag(const half_t* X, int stride, int m0, int k0, int lane) {
    const half_t* p = X + (m0 + (lane & 15)) * stride + k0 + ((lane >> 4) << 3);
    v16h a;
#pragma unroll
    for (int i = 0; i < 8; ++i) { a[i] = p[i]; a[8 + i] = p[16 + i]; }
    return a;
}

// B: 32x16 (KxN), weights stored transposed in LDS as Wt[N][K] row-major.
// lanes 0-15 hold K k0..k0+7, k0+16..23 of column n, lanes 16-31 hold k0+8..15, k0+24..31.
__device__ inline v16h load_b_frag(const half_t* Wt, int K, int n0, int k0, int lane) {
    const half_t* p = Wt + (n0 + (lane & 15)) * K + k0 + ((lane >> 4) << 3);
    v16h b;
#pragma unroll
    for (int i = 0; i < 8; ++i) { b[i] = p[i]; b[8 + i] = p[16 + i]; }
    return b;
}

// One wave computes a 16-row strip of a 64xN output:  Out = act((X @ W) * cs + ct)
__device__ inline void wave_gemm(const half_t* X, int xstride, const half_t* Wt,
                                 int K, int N, const float* cs, const float* ct,
                                 bool do_relu, half_t* Out, int ostride, int ocol,
                                 int wid, int lane) {
    const int m0 = wid * 16;
    for (int n0 = 0; n0 < N; n0 += 16) {
        v8f acc = {};
        for (int k0 = 0; k0 < K; k0 += 32) {
            v16h a = load_a_frag(X, xstride, m0, k0, lane);
            v16h b = load_b_frag(Wt, K, n0, k0, lane);
            acc = __builtin_amdgcn_wmma_f32_16x16x32_f16(false, a, false, b,
                                                         (short)0, acc, false, false);
        }
        const int n = n0 + (lane & 15);
        const int rb = m0 + ((lane >> 4) << 3);
        const float s = cs[n], t = ct[n];
#pragma unroll
        for (int v = 0; v < 8; ++v) {
            float val = acc[v] * s + t;
            if (do_relu) val = fmaxf(val, 0.0f);
            Out[(rb + v) * ostride + ocol + n] = (half_t)val;
        }
    }
}

// ---------------------------------------------------------------------------
// Weight tile load: pre-transposed f16 weights in global -> LDS.
// Preferred path: Tensor Data Mover, 1-D D# descriptor, issued by wave 0 only
// (TDM ignores EXEC; one instruction moves the whole tile). Fallback: plain
// vectorized copy.
__device__ inline void load_wt_issue(const half_t* __restrict__ gsrc, half_t* Wt,
                                     int nhalves, int tid, int wid) {
#if __has_builtin(__builtin_amdgcn_tensor_load_to_lds)
    if (wid == 0) {
        unsigned lds = (unsigned)(uintptr_t)(void*)Wt;      // LDS generic addr low bits
        unsigned long long ga = (unsigned long long)(uintptr_t)gsrc;
        unsigned L = (unsigned)nhalves;                     // elements (data_size = 2B)
        v4u g0;
        g0[0] = 1u;                                          // count=1, user D#
        g0[1] = lds;                                         // lds_addr
        g0[2] = (unsigned)(ga & 0xffffffffu);                // global_addr[31:0]
        g0[3] = (unsigned)((ga >> 32) & 0x01ffffffu) | (2u << 30); // addr[56:32] | type=2
        v8i g1;
        g1[0] = (int)(1u << 16);                             // data_size=1 -> 2 bytes
        g1[1] = (int)((L & 0xffffu) << 16);                  // tensor_dim0 lo16 (bits 63:48)
        g1[2] = (int)(((L >> 16) & 0xffffu) | (1u << 16));   // tensor_dim0 hi | tensor_dim1=1
        g1[3] = (int)((L & 0xffffu) << 16);                  // tile_dim0 = L (bits 127:112)
        g1[4] = 0;                                           // tile_dim1=0, tile_dim2=0
        g1[5] = (int)L;                                      // tensor_dim0_stride lo32
        g1[6] = 0;
        g1[7] = 0;
        v4i g2 = {0, 0, 0, 0};
        v4i g3 = {0, 0, 0, 0};
        v8i g4 = {0, 0, 0, 0, 0, 0, 0, 0};                   // unused second block
        __builtin_amdgcn_tensor_load_to_lds(g0, g1, g2, g3, g4, 0);
    }
    (void)tid;
#else
    const v8h* s = (const v8h*)gsrc;
    v8h* d = (v8h*)Wt;
    for (int e = tid; e < (nhalves >> 3); e += NTHR) d[e] = s[e];
    (void)wid;
#endif
}

__device__ inline void wt_wait(int wid) {
#if __has_builtin(__builtin_amdgcn_tensor_load_to_lds)
    if (wid == 0) __builtin_amdgcn_s_wait_tensorcnt(0);
#else
    (void)wid;
#endif
}

// Fold bias + optional eval-mode BatchNorm into per-column scale/shift.
__device__ inline void load_colparams(const float* bias, const float* bn_g,
                                      const float* bn_b, int N,
                                      float* cs, float* ct, int tid) {
    if (tid < N) {
        float s = 1.0f, t = 0.0f;
        if (bn_g) { s = bn_g[tid] * rsqrtf(1.0f + 1e-5f); t = bn_b[tid]; }
        cs[tid] = s;
        ct[tid] = bias[tid] * s + t;
    }
}

// ---------------------------------------------------------------------------
// One-time prep: convert f32 W[K][N] -> f16 transposed Wt[N][K] in workspace.
struct WJob { const float* src; int K; int N; int off; };
struct WJobs { WJob j[17]; };

__global__ __launch_bounds__(256)
void prep_weights_kernel(WJobs jobs, half_t* __restrict__ dst) {
    const WJob jb = jobs.j[blockIdx.x];
    const int KN = jb.K * jb.N;
    for (int e = threadIdx.x; e < KN; e += 256) {
        int n = e / jb.K, k = e - n * jb.K;
        dst[jb.off + e] = (half_t)jb.src[(size_t)k * jb.N + n];
    }
}

// ---------------------------------------------------------------------------
// Aggregator: one workgroup per batch element b.
template <bool HAS_WR>
__global__ __launch_bounds__(NTHR)
void aggr_kernel(const int* __restrict__ self_idx,
                 const int* __restrict__ hist,
                 const int* __restrict__ histR,
                 const float* __restrict__ embSelf,
                 const float* __restrict__ embN,
                 const float* __restrict__ r2e,
                 const half_t* __restrict__ wr1t, const float* __restrict__ wr1_b,
                 const half_t* __restrict__ wr2t, const float* __restrict__ wr2_b,
                 const half_t* __restrict__ a1t,  const float* __restrict__ a1_b,
                 const half_t* __restrict__ a2t,  const float* __restrict__ a2_b,
                 const float* __restrict__ a3_w,  const float* __restrict__ a3_b,
                 const float* __restrict__ lin1_w, const float* __restrict__ lin1_b,
                 float* __restrict__ out) {
    __shared__ half_t X[64 * 128];
    __shared__ half_t Hb[64 * 64];
    __shared__ half_t Ob[64 * 64];
    __shared__ half_t Wt[64 * 128];
    __shared__ float selfv[64], cs[64], ct[64], lg[64], att[64], cat[128], red[2];

    const int b = blockIdx.x;
    const int tid = threadIdx.x, lane = tid & 31, wid = tid >> 5;

    // kick off the first weight DMA so it overlaps the gather
    load_wt_issue(HAS_WR ? wr1t : a1t, Wt, 128 * 64, tid, wid);

    if (tid < 64) selfv[tid] = embSelf[(size_t)self_idx[b] * 64 + tid];

    // ---- gather neighbor embeddings (2 threads per row, 32 cols each)
    {
        const int r = tid >> 1, c0 = (tid & 1) * 32;
        const int idx = hist[(size_t)b * 64 + r];
        if (HAS_WR) {
            const int ridx = histR[(size_t)b * 64 + r];
            for (int c = c0; c < c0 + 32; ++c) {
                X[r * 128 + c]      = (half_t)embN[(size_t)idx * 64 + c];
                X[r * 128 + 64 + c] = (half_t)r2e[(size_t)ridx * 64 + c];
            }
        } else {
            for (int c = c0; c < c0 + 32; ++c)
                Ob[r * 64 + c] = (half_t)embN[(size_t)idx * 64 + c];
        }
    }

    if (HAS_WR) {
        load_colparams(wr1_b, nullptr, nullptr, 64, cs, ct, tid);
        wt_wait(wid);
        __syncthreads();
        wave_gemm(X, 128, Wt, 128, 64, cs, ct, true, Hb, 64, 0, wid, lane);
        __syncthreads();
        load_wt_issue(wr2t, Wt, 64 * 64, tid, wid);
        load_colparams(wr2_b, nullptr, nullptr, 64, cs, ct, tid);
        wt_wait(wid);
        __syncthreads();
        wave_gemm(Hb, 64, Wt, 64, 64, cs, ct, true, Ob, 64, 0, wid, lane);
        __syncthreads();
        load_wt_issue(a1t, Wt, 128 * 64, tid, wid);   // prefetch attention weights
    }

    if (!HAS_WR) __syncthreads();   // Ob + selfv ready

    // ---- attention input: [O | self]
    for (int e = tid; e < 64 * 64; e += NTHR) {
        const int r = e >> 6, c = e & 63;
        X[r * 128 + c]      = Ob[e];
        X[r * 128 + 64 + c] = (half_t)selfv[c];
    }
    load_colparams(a1_b, nullptr, nullptr, 64, cs, ct, tid);
    wt_wait(wid);
    __syncthreads();
    wave_gemm(X, 128, Wt, 128, 64, cs, ct, true, Hb, 64, 0, wid, lane);
    __syncthreads();
    load_wt_issue(a2t, Wt, 64 * 64, tid, wid);
    load_colparams(a2_b, nullptr, nullptr, 64, cs, ct, tid);
    wt_wait(wid);
    __syncthreads();
    wave_gemm(Hb, 64, Wt, 64, 64, cs, ct, true, X, 64, 0, wid, lane);  // A2 reuses X
    __syncthreads();

    // ---- logits + softmax over the 64 neighbors
    if (tid < 64) {
        float acc = a3_b[0];
        for (int k = 0; k < 64; ++k) acc += (float)X[tid * 64 + k] * a3_w[k];
        lg[tid] = acc;
    }
    __syncthreads();
    if (tid == 0) { float m = lg[0]; for (int i = 1; i < 64; ++i) m = fmaxf(m, lg[i]); red[0] = m; }
    __syncthreads();
    if (tid < 64) att[tid] = __expf(lg[tid] - red[0]);
    __syncthreads();
    if (tid == 0) { float s = 0.f; for (int i = 0; i < 64; ++i) s += att[i]; red[1] = 1.0f / s; }
    __syncthreads();

    // ---- neigh = sum(att * O), then out = relu([self|neigh] @ lin1 + b)
    if (tid < 64) {
        const float inv = red[1];
        float acc = 0.f;
        for (int r = 0; r < 64; ++r) acc += att[r] * inv * (float)Ob[r * 64 + tid];
        cat[tid] = selfv[tid];
        cat[64 + tid] = acc;
    }
    __syncthreads();
    if (tid < 64) {
        float acc = lin1_b[tid];
        for (int k = 0; k < 128; ++k) acc += cat[k] * lin1_w[(size_t)k * 64 + tid];
        out[(size_t)b * 64 + tid] = fmaxf(acc, 0.0f);
    }
}

// ---------------------------------------------------------------------------
// Top MLP: one workgroup per 64 batch rows; whole chain via WMMA with BN folded.
__global__ __launch_bounds__(NTHR)
void top_kernel(const float* __restrict__ item_space,
                const float* __restrict__ social_space,
                const float* __restrict__ item_latent,
                const half_t* __restrict__ wut,   const float* __restrict__ wu_b,
                const half_t* __restrict__ wur1t, const float* __restrict__ wur1_b,
                const half_t* __restrict__ wur2t, const float* __restrict__ wur2_b,
                const half_t* __restrict__ wir1t, const float* __restrict__ wir1_b,
                const half_t* __restrict__ wir2t, const float* __restrict__ wir2_b,
                const half_t* __restrict__ wui1t, const float* __restrict__ wui1_b,
                const half_t* __restrict__ wui2t, const float* __restrict__ wui2_b,
                const float* __restrict__ wui3_w, const float* __restrict__ wui3_b,
                const float* __restrict__ bn1_g, const float* __restrict__ bn1_b,
                const float* __restrict__ bn2_g, const float* __restrict__ bn2_b,
                const float* __restrict__ bn3_g, const float* __restrict__ bn3_b,
                const float* __restrict__ bn4_g, const float* __restrict__ bn4_b,
                float* __restrict__ out) {
    __shared__ half_t X[64 * 128];
    __shared__ half_t T1[64 * 64];
    __shared__ half_t T2[64 * 64];
    __shared__ half_t Wt[64 * 128];
    __shared__ float cs[64], ct[64];

    const int r0 = blockIdx.x * 64;
    const int tid = threadIdx.x, lane = tid & 31, wid = tid >> 5;

    load_wt_issue(wut, Wt, 128 * 64, tid, wid);

    // X = [item_space | social_space]
    for (int e = tid; e < 64 * 64; e += NTHR) {
        const int r = e >> 6, c = e & 63;
        X[r * 128 + c]      = (half_t)item_space[(size_t)(r0 + r) * 64 + c];
        X[r * 128 + 64 + c] = (half_t)social_space[(size_t)(r0 + r) * 64 + c];
    }
    load_colparams(wu_b, nullptr, nullptr, 64, cs, ct, tid);
    wt_wait(wid);
    __syncthreads();
    wave_gemm(X, 128, Wt, 128, 64, cs, ct, true, T1, 64, 0, wid, lane);
    __syncthreads();
    load_wt_issue(wur1t, Wt, 64 * 64, tid, wid);
    load_colparams(wur1_b, bn1_g, bn1_b, 64, cs, ct, tid);
    wt_wait(wid);
    __syncthreads();
    wave_gemm(T1, 64, Wt, 64, 64, cs, ct, true, T2, 64, 0, wid, lane);
    __syncthreads();
    load_wt_issue(wur2t, Wt, 64 * 64, tid, wid);
    load_colparams(wur2_b, nullptr, nullptr, 64, cs, ct, tid);
    wt_wait(wid);
    __syncthreads();
    wave_gemm(T2, 64, Wt, 64, 64, cs, ct, false, X, 128, 0, wid, lane);  // u_lat -> X[:,0:64]
    __syncthreads();

    load_wt_issue(wir1t, Wt, 64 * 64, tid, wid);
    for (int e = tid; e < 64 * 64; e += NTHR) {
        const int r = e >> 6, c = e & 63;
        T1[e] = (half_t)item_latent[(size_t)(r0 + r) * 64 + c];
    }
    load_colparams(wir1_b, bn2_g, bn2_b, 64, cs, ct, tid);
    wt_wait(wid);
    __syncthreads();
    wave_gemm(T1, 64, Wt, 64, 64, cs, ct, true, T2, 64, 0, wid, lane);
    __syncthreads();
    load_wt_issue(wir2t, Wt, 64 * 64, tid, wid);
    load_colparams(wir2_b, nullptr, nullptr, 64, cs, ct, tid);
    wt_wait(wid);
    __syncthreads();
    wave_gemm(T2, 64, Wt, 64, 64, cs, ct, false, X, 128, 64, wid, lane); // i_lat -> X[:,64:128]
    __syncthreads();

    load_wt_issue(wui1t, Wt, 128 * 64, tid, wid);
    load_colparams(wui1_b, bn3_g, bn3_b, 64, cs, ct, tid);
    wt_wait(wid);
    __syncthreads();
    wave_gemm(X, 128, Wt, 128, 64, cs, ct, true, T1, 64, 0, wid, lane);
    __syncthreads();
    load_wt_issue(wui2t, Wt, 64 * 16, tid, wid);
    load_colparams(wui2_b, bn4_g, bn4_b, 16, cs, ct, tid);
    wt_wait(wid);
    __syncthreads();
    wave_gemm(T1, 64, Wt, 64, 16, cs, ct, true, T2, 16, 0, wid, lane);
    __syncthreads();

    if (tid < 64) {
        float acc = wui3_b[0];
        for (int k = 0; k < 16; ++k) acc += (float)T2[tid * 16 + k] * wui3_w[k];
        out[r0 + tid] = acc;
    }
}

// ---------------------------------------------------------------------------
extern "C" void kernel_launch(void* const* d_in, const int* in_sizes, int n_in,
                              void* d_out, int out_size, void* d_ws, size_t ws_size,
                              hipStream_t stream) {
    (void)n_in; (void)out_size; (void)ws_size;
    const int B = in_sizes[0];

    const int* nodes_u      = (const int*)d_in[0];
    const int* nodes_i      = (const int*)d_in[1];
    const int* hist_u_items = (const int*)d_in[2];
    const int* hist_u_r     = (const int*)d_in[3];
    const int* hist_i_users = (const int*)d_in[4];
    const int* hist_i_r     = (const int*)d_in[5];
    const int* social_neigh = (const int*)d_in[6];

#define F(i) ((const float*)d_in[i])
    const float *u2e = F(7), *i2e = F(8), *r2e = F(9);
    const float *u_wr1_w = F(10), *u_wr1_b = F(11), *u_wr2_w = F(12), *u_wr2_b = F(13);
    const float *u_a1_w = F(14), *u_a1_b = F(15), *u_a2_w = F(16), *u_a2_b = F(17);
    const float *u_a3_w = F(18), *u_a3_b = F(19), *u_lin1_w = F(20), *u_lin1_b = F(21);
    const float *i_wr1_w = F(22), *i_wr1_b = F(23), *i_wr2_w = F(24), *i_wr2_b = F(25);
    const float *i_a1_w = F(26), *i_a1_b = F(27), *i_a2_w = F(28), *i_a2_b = F(29);
    const float *i_a3_w = F(30), *i_a3_b = F(31), *i_lin1_w = F(32), *i_lin1_b = F(33);
    const float *s_a1_w = F(34), *s_a1_b = F(35), *s_a2_w = F(36), *s_a2_b = F(37);
    const float *s_a3_w = F(38), *s_a3_b = F(39), *s_lin1_w = F(40), *s_lin1_b = F(41);
    const float *wu_w = F(42), *wu_b = F(43);
    const float *wur1_w = F(44), *wur1_b = F(45), *wur2_w = F(46), *wur2_b = F(47);
    const float *wir1_w = F(48), *wir1_b = F(49), *wir2_w = F(50), *wir2_b = F(51);
    const float *wui1_w = F(52), *wui1_b = F(53);
    const float *wui2_w = F(54), *wui2_b = F(55);
    const float *wui3_w = F(56), *wui3_b = F(57);
    const float *bn1_g = F(58), *bn1_b = F(59), *bn2_g = F(60), *bn2_b = F(61);
    const float *bn3_g = F(62), *bn3_b = F(63), *bn4_g = F(64), *bn4_b = F(65);
#undef F

    float* item_space  = (float*)d_ws;                       // [B,64]
    float* social_sp   = item_space + (size_t)B * 64;        // [B,64]
    float* item_latent = social_sp + (size_t)B * 64;         // [B,64]
    half_t* wbase      = (half_t*)(item_latent + (size_t)B * 64);

    // f16 transposed weight layout (offsets in halves)
    int off = 0;
    WJobs jobs;
    int oj = 0;
    auto add = [&](const float* src, int K, int N) {
        jobs.j[oj].src = src; jobs.j[oj].K = K; jobs.j[oj].N = N; jobs.j[oj].off = off;
        int o = off; off += K * N; ++oj; return o;
    };
    const int o_uwr1 = add(u_wr1_w, 128, 64), o_uwr2 = add(u_wr2_w, 64, 64);
    const int o_ua1  = add(u_a1_w, 128, 64),  o_ua2  = add(u_a2_w, 64, 64);
    const int o_iwr1 = add(i_wr1_w, 128, 64), o_iwr2 = add(i_wr2_w, 64, 64);
    const int o_ia1  = add(i_a1_w, 128, 64),  o_ia2  = add(i_a2_w, 64, 64);
    const int o_sa1  = add(s_a1_w, 128, 64),  o_sa2  = add(s_a2_w, 64, 64);
    const int o_wu   = add(wu_w, 128, 64);
    const int o_wur1 = add(wur1_w, 64, 64),   o_wur2 = add(wur2_w, 64, 64);
    const int o_wir1 = add(wir1_w, 64, 64),   o_wir2 = add(wir2_w, 64, 64);
    const int o_wui1 = add(wui1_w, 128, 64);
    const int o_wui2 = add(wui2_w, 64, 16);

    dim3 blk(NTHR);

    prep_weights_kernel<<<dim3(17), dim3(256), 0, stream>>>(jobs, wbase);

    // enc_u: user's item history
    aggr_kernel<true><<<dim3(B), blk, 0, stream>>>(
        nodes_u, hist_u_items, hist_u_r, u2e, i2e, r2e,
        wbase + o_uwr1, u_wr1_b, wbase + o_uwr2, u_wr2_b,
        wbase + o_ua1, u_a1_b, wbase + o_ua2, u_a2_b,
        u_a3_w, u_a3_b, u_lin1_w, u_lin1_b, item_space);

    // social: user's social neighbors (no wr MLP)
    aggr_kernel<false><<<dim3(B), blk, 0, stream>>>(
        nodes_u, social_neigh, nullptr, u2e, u2e, r2e,
        nullptr, nullptr, nullptr, nullptr,
        wbase + o_sa1, s_a1_b, wbase + o_sa2, s_a2_b,
        s_a3_w, s_a3_b, s_lin1_w, s_lin1_b, social_sp);

    // enc_i: item's user history
    aggr_kernel<true><<<dim3(B), blk, 0, stream>>>(
        nodes_i, hist_i_users, hist_i_r, i2e, u2e, r2e,
        wbase + o_iwr1, i_wr1_b, wbase + o_iwr2, i_wr2_b,
        wbase + o_ia1, i_a1_b, wbase + o_ia2, i_a2_b,
        i_a3_w, i_a3_b, i_lin1_w, i_lin1_b, item_latent);

    // top MLP -> scores [B]
    top_kernel<<<dim3(B / 64), blk, 0, stream>>>(
        item_space, social_sp, item_latent,
        wbase + o_wu, wu_b, wbase + o_wur1, wur1_b, wbase + o_wur2, wur2_b,
        wbase + o_wir1, wir1_b, wbase + o_wir2, wir2_b,
        wbase + o_wui1, wui1_b, wbase + o_wui2, wui2_b,
        wui3_w, wui3_b,
        bn1_g, bn1_b, bn2_g, bn2_b, bn3_g, bn3_b, bn4_g, bn4_b,
        (float*)d_out);
}
